// RetrievalPolicyTriple2_73065983640362
// MI455X (gfx1250) — matrix operand order repaired and checked by reading, not compiled, not run on hardware
//
#include <hip/hip_runtime.h>
#include <cmath>
#include <cstdint>
#include <cstddef>

#define NN 50000
#define EE 250000
#define DD 768
#define HH 256
#define ZMIX 0.8f
#define NSLOPE 0.2f

typedef __attribute__((ext_vector_type(16))) __bf16 v16bf;
typedef __attribute__((ext_vector_type(8)))  float v8f;

union FragAB { v16bf v; unsigned int u[8]; };
union FragC  { v8f  v; float f[8]; };

// round-half-up bf16 conversion: 1 add + 1 shift (cheap, co-executes with WMMA)
__device__ __forceinline__ unsigned short f2bf(float x) {
  return (unsigned short)((__float_as_uint(x) + 0x8000u) >> 16);
}

__device__ __forceinline__ unsigned int pk2(float a, float b) {
  return ((__float_as_uint(a) + 0x8000u) >> 16) |
         ((__float_as_uint(b) + 0x8000u) & 0xFFFF0000u);
}

__device__ __forceinline__ float lrelu(float x) { return x > 0.f ? x : NSLOPE * x; }

__device__ __forceinline__ float waveRed(float v) {
#pragma unroll
  for (int o = 16; o > 0; o >>= 1) v += __shfl_xor(v, o, 32);
  return v;
}

__device__ __forceinline__ void atomicMaxF(float* a, float v) {
  if (!(v < 0.f)) atomicMax((int*)a, __float_as_int(v));
  else            atomicMin((unsigned int*)a, (unsigned int)__float_as_int(v));
}

// ---------------------------------------------------------------------------
// WMMA GEMM: out[M x 256] = act( A(MODE)[M x K] @ W[K x 256] + bias )
// MODE: 0 plain(A0, lda)            1 concat(A0[M x 256], A1 = q[256])
//       2 concat(A0[Mx256], A1[Mx256])
//       3 triple: concat(X[gs[m]], A1[m], X[gd[m]])  (X = A0, N x 256)
// Block: 256 threads (8 waves). Tile: 128(M) x 64(N) x 32(K) per step.
// Double-buffered LDS; next tile's global loads issued before current WMMAs.
// ---------------------------------------------------------------------------
template <int MODE>
__device__ __forceinline__ void loadAtile(float4 (&ra)[4], int k0, int row0, int M, int lda,
                                          const float* __restrict__ A0,
                                          const float* __restrict__ A1,
                                          const int* __restrict__ s_src,
                                          const int* __restrict__ s_dst, int tid) {
#pragma unroll
  for (int it = 0; it < 4; ++it) {
    int g  = tid + it * 256;             // 1024 groups of 4 k-elements
    int r  = g >> 3;
    int kc = (g & 7) << 2;
    int grow = row0 + r;
    float4 v4 = make_float4(0.f, 0.f, 0.f, 0.f);
    if (grow < M) {
      int k = k0 + kc;                   // 4-aligned; never crosses a 256-seg
      const float* p;
      if (MODE == 0)      p = &A0[(size_t)grow * lda + k];
      else if (MODE == 1) p = (k < HH) ? &A0[(size_t)grow * HH + k]
                                       : &A1[k - HH];
      else if (MODE == 2) p = (k < HH) ? &A0[(size_t)grow * HH + k]
                                       : &A1[(size_t)grow * HH + (k - HH)];
      else {
        if (k < HH)          p = &A0[(size_t)s_src[r] * HH + k];
        else if (k < 2 * HH) p = &A1[(size_t)grow * HH + (k - HH)];
        else                 p = &A0[(size_t)s_dst[r] * HH + (k - 2 * HH)];
      }
      v4 = *(const float4*)p;
    }
    ra[it] = v4;
  }
}

__device__ __forceinline__ void storeAtile(unsigned short* __restrict__ sa,
                                           const float4 (&ra)[4], int tid) {
#pragma unroll
  for (int it = 0; it < 4; ++it) {
    int g  = tid + it * 256;
    int r  = g >> 3;
    int kc = (g & 7) << 2;
    *(uint2*)&sa[r * 32 + kc] = make_uint2(pk2(ra[it].x, ra[it].y),
                                           pk2(ra[it].z, ra[it].w));
  }
}

__device__ __forceinline__ void loadBtile(float4 (&rb)[2], int k0, int n0,
                                          const float* __restrict__ W, int tid) {
#pragma unroll
  for (int it = 0; it < 2; ++it) {
    int g  = tid + it * 256;             // 512 groups of 4 n-elements
    int kk = g >> 4;
    int nn = (g & 15) << 2;
    rb[it] = *(const float4*)&W[(size_t)(k0 + kk) * HH + (n0 + nn)];
  }
}

__device__ __forceinline__ void storeBtile(unsigned short* __restrict__ sb,
                                           const float4 (&rb)[2], int tid) {
#pragma unroll
  for (int it = 0; it < 2; ++it) {
    int g  = tid + it * 256;
    int kk = g >> 4;
    int nn = (g & 15) << 2;
    sb[(nn + 0) * 32 + kk] = f2bf(rb[it].x);
    sb[(nn + 1) * 32 + kk] = f2bf(rb[it].y);
    sb[(nn + 2) * 32 + kk] = f2bf(rb[it].z);
    sb[(nn + 3) * 32 + kk] = f2bf(rb[it].w);
  }
}

template <int MODE>
__global__ void __launch_bounds__(256)
k_gemm(int M, int K, int lda,
       const float* __restrict__ A0, const float* __restrict__ A1,
       const int* __restrict__ gs, const int* __restrict__ gd,
       const float* __restrict__ W, const float* __restrict__ bias,
       int relu, float* __restrict__ out) {
  __shared__ unsigned short sa[2][128 * 32];   // A tiles, row-major (m, k), bf16
  __shared__ unsigned short sb[2][64 * 32];    // B tiles, transposed (n, k), bf16
  __shared__ int s_src[128], s_dst[128];       // row gather indices (MODE 3)

  const int tid  = threadIdx.x;
  const int row0 = blockIdx.x * 128;
  const int n0   = blockIdx.y * 64;

  if (MODE == 3) {
    if (tid < 128) {
      int grow = row0 + tid;
      s_src[tid] = (grow < M) ? gs[grow] : 0;
      s_dst[tid] = (grow < M) ? gd[grow] : 0;
    }
    __syncthreads();
  }

  FragC acc[4];
#pragma unroll
  for (int g = 0; g < 4; ++g)
#pragma unroll
    for (int r = 0; r < 8; ++r) acc[g].f[r] = 0.f;

  const int lane = tid & 31;
  const int wv   = tid >> 5;
  const int lr   = lane & 15;
  const int kb   = (lane < 16) ? 0 : 8;    // unused; A fragment uses koff
  const int koff = (lane < 16) ? 0 : 16;   // B-fragment K base

  // ---- prologue: stage tile 0 ----
  float4 ra[4], rb[2];
  loadAtile<MODE>(ra, 0, row0, M, lda, A0, A1, s_src, s_dst, tid);
  loadBtile(rb, 0, n0, W, tid);
  storeAtile(sa[0], ra, tid);
  storeBtile(sb[0], rb, tid);
  __syncthreads();

  int p = 0;
  for (int k0 = 0; k0 < K; k0 += 32) {
    const bool has_next = (k0 + 32 < K);
    // ---- issue next tile's global loads (latency hidden behind WMMAs) ----
    if (has_next) {
      loadAtile<MODE>(ra, k0 + 32, row0, M, lda, A0, A1, s_src, s_dst, tid);
      loadBtile(rb, k0 + 32, n0, W, tid);
    }

    // ---- compute current tile from LDS buffer p ----
    const unsigned short* cap = sa[p];
    const unsigned short* cbp = sb[p];
    FragAB af;
#pragma unroll
    for (int v = 0; v < 8; ++v)
      af.u[v] = *(const unsigned int*)&cap[(wv * 16 + lr) * 32 + koff + 2 * v];

    FragAB bf_[4];
#pragma unroll
    for (int g = 0; g < 4; ++g)
#pragma unroll
      for (int v = 0; v < 8; ++v)
        bf_[g].u[v] = *(const unsigned int*)&cbp[(g * 16 + lr) * 32 + koff + 2 * v];

#pragma unroll
    for (int g = 0; g < 4; ++g)
      acc[g].v = __builtin_amdgcn_wmma_f32_16x16x32_bf16(
          false, af.v, false, bf_[g].v, (short)0, acc[g].v, false, false);

    // ---- convert + store next tile into the other buffer ----
    if (has_next) {
      storeAtile(sa[p ^ 1], ra, tid);
      storeBtile(sb[p ^ 1], rb, tid);
    }
    __syncthreads();
    p ^= 1;
  }

  // ---- epilogue: C/D layout (VGPR r: M=r | M=8+r; N=lane%16) ----
  const int mhalf = (lane < 16) ? 0 : 8;
#pragma unroll
  for (int g = 0; g < 4; ++g) {
    int col = n0 + g * 16 + lr;
    float bv = bias ? bias[col] : 0.f;
#pragma unroll
    for (int r = 0; r < 8; ++r) {
      int m = row0 + wv * 16 + mhalf + r;
      if (m < M) {
        float v = acc[g].f[r] + bv;
        if (relu) v = v > 0.f ? v : 0.f;
        out[(size_t)m * HH + col] = v;
      }
    }
  }
}

// ---------------------------------------------------------------------------
// Support kernels
// ---------------------------------------------------------------------------
__global__ void k_fill(float* p, float v, size_t n) {
  size_t i = (size_t)blockIdx.x * blockDim.x + threadIdx.x;
  if (i < n) p[i] = v;
}

__global__ void __launch_bounds__(256)
k_rowatt(const float* __restrict__ h, const float* __restrict__ asrc,
         const float* __restrict__ adst, float* __restrict__ ssum,
         float* __restrict__ dsum, int n) {
  int wv = threadIdx.x >> 5, lane = threadIdx.x & 31;
  int row = blockIdx.x * 8 + wv;
  if (row >= n) return;
  const float* hr = h + (size_t)row * HH;
  float s = 0.f, d = 0.f;
  for (int i = lane; i < HH; i += 32) { float hv = hr[i]; s += hv * asrc[i]; d += hv * adst[i]; }
  s = waveRed(s); d = waveRed(d);
  if (lane == 0) { ssum[row] = s; dsum[row] = d; }
}

__global__ void k_edge_a_max(const float* __restrict__ ssum, const float* __restrict__ dsum,
                             const int* __restrict__ s, const int* __restrict__ d,
                             float* __restrict__ a, float* __restrict__ m, int E) {
  int e = blockIdx.x * blockDim.x + threadIdx.x; if (e >= E) return;
  float v = lrelu(ssum[s[e]] + dsum[d[e]]);
  a[e] = v; atomicMaxF(&m[d[e]], v);
}

__global__ void k_node_self_max(const float* __restrict__ ssum, const float* __restrict__ dsum,
                                float* __restrict__ m, int n) {
  int i = blockIdx.x * blockDim.x + threadIdx.x; if (i >= n) return;
  atomicMaxF(&m[i], lrelu(ssum[i] + dsum[i]));
}

__global__ void k_edge_exp(const float* __restrict__ a, const float* __restrict__ m,
                           const int* __restrict__ d, float* __restrict__ ea,
                           float* __restrict__ denom, int E) {
  int e = blockIdx.x * blockDim.x + threadIdx.x; if (e >= E) return;
  float v = expf(a[e] - m[d[e]]);
  ea[e] = v; atomicAdd(&denom[d[e]], v);
}

__global__ void k_node_self_exp(const float* __restrict__ ssum, const float* __restrict__ dsum,
                                const float* __restrict__ m, float* __restrict__ denom, int n) {
  int i = blockIdx.x * blockDim.x + threadIdx.x; if (i >= n) return;
  atomicAdd(&denom[i], expf(lrelu(ssum[i] + dsum[i]) - m[i]));
}

__global__ void __launch_bounds__(256)
k_edge_scatter(const float* __restrict__ ea, const float* __restrict__ denom,
               const int* __restrict__ s, const int* __restrict__ d,
               const float* __restrict__ h, float* __restrict__ out, int E) {
  int e = blockIdx.x; if (e >= E) return;
  int j = threadIdx.x;
  int ds = d[e], ss = s[e];
  float alpha = ea[e] / (denom[ds] + 1e-16f);
  atomicAdd(&out[(size_t)ds * HH + j], alpha * h[(size_t)ss * HH + j]);
}

__global__ void __launch_bounds__(256)
k_node_self_scatter(const float* __restrict__ ssum, const float* __restrict__ dsum,
                    const float* __restrict__ m, const float* __restrict__ denom,
                    const float* __restrict__ h, float* __restrict__ out) {
  int i = blockIdx.x; int j = threadIdx.x;
  float alpha = expf(lrelu(ssum[i] + dsum[i]) - m[i]) / (denom[i] + 1e-16f);
  out[(size_t)i * HH + j] += alpha * h[(size_t)i * HH + j];
}

__global__ void k_add_bias(float* __restrict__ x, const float* __restrict__ b, int tot) {
  int idx = blockIdx.x * blockDim.x + threadIdx.x; if (idx >= tot) return;
  x[idx] += b[idx & (HH - 1)];
}

__global__ void k_mix(const unsigned char* __restrict__ mask, const float* __restrict__ x1,
                      const float* __restrict__ x0, float* __restrict__ out, int M) {
  int idx = blockIdx.x * blockDim.x + threadIdx.x; if (idx >= M * HH) return;
  int i = idx >> 8;
  float a = x1[idx], b = x0[idx];
  out[idx] = mask[i] ? (ZMIX * a + (1.f - ZMIX) * b) : (ZMIX * b + (1.f - ZMIX) * a);
}

__global__ void __launch_bounds__(256)
k_colsum(const float* __restrict__ x, float* __restrict__ acc, int M) {
  int r0 = blockIdx.x * 64; int j = threadIdx.x;
  float s = 0.f;
  for (int r = 0; r < 64; ++r) { int row = r0 + r; if (row < M) s += x[(size_t)row * HH + j]; }
  atomicAdd(&acc[j], s);
}

__global__ void __launch_bounds__(256)
k_colvar(const float* __restrict__ x, const float* __restrict__ mean,
         const float* __restrict__ ms, float* __restrict__ acc, int M) {
  int r0 = blockIdx.x * 64; int j = threadIdx.x;
  float c = ms[j] * mean[j], s = 0.f;
  for (int r = 0; r < 64; ++r) {
    int row = r0 + r;
    if (row < M) { float xc = x[(size_t)row * HH + j] - c; s += xc * xc; }
  }
  atomicAdd(&acc[j], s);
}

__global__ void k_finalize(const float* __restrict__ src, float* __restrict__ dst,
                           float M, float eps, int isrstd) {
  int j = threadIdx.x;
  float v = src[j] / M;
  dst[j] = isrstd ? rsqrtf(v + eps) : v;
}

__global__ void k_gn_apply(const float* __restrict__ x, const float* __restrict__ mean,
                           const float* __restrict__ rstd, const float* __restrict__ ms,
                           const float* __restrict__ w, const float* __restrict__ b,
                           float* __restrict__ out, int M) {
  int idx = blockIdx.x * blockDim.x + threadIdx.x; if (idx >= M * HH) return;
  int j = idx & (HH - 1);
  float xc = x[idx] - ms[j] * mean[j];
  out[idx] = w[j] * xc * rstd[j] + b[j];
}

__global__ void __launch_bounds__(256)
k_dot_head(const float* __restrict__ hid, const float* __restrict__ w,
           const float* __restrict__ b, float* __restrict__ out, int M) {
  int wv = threadIdx.x >> 5, lane = threadIdx.x & 31;
  int row = blockIdx.x * 8 + wv; if (row >= M) return;
  const float* hr = hid + (size_t)row * HH;
  float s = 0.f;
  for (int i = lane; i < HH; i += 32) s += hr[i] * w[i];
  s = waveRed(s);
  if (lane == 0) out[row] = s + b[0];
}

__global__ void k_logit_post(float* __restrict__ l, const unsigned char* __restrict__ am,
                             const float* __restrict__ ab, int E) {
  int i = blockIdx.x * blockDim.x + threadIdx.x; if (i >= E) return;
  float v = am[i] ? l[i] : -1e9f;
  l[i] = v + logf(ab[i] + 1e-10f);
}

__global__ void __launch_bounds__(256)
k_redmax(const float* __restrict__ l, float* __restrict__ red, int E) {
  __shared__ float sh[256];
  int i = blockIdx.x * 256 + threadIdx.x;
  sh[threadIdx.x] = (i < E) ? l[i] : -INFINITY;
  __syncthreads();
  for (int st = 128; st > 0; st >>= 1) {
    if (threadIdx.x < st) sh[threadIdx.x] = fmaxf(sh[threadIdx.x], sh[threadIdx.x + st]);
    __syncthreads();
  }
  if (threadIdx.x == 0) atomicMaxF(&red[0], sh[0]);
}

__global__ void __launch_bounds__(256)
k_redsum(const float* __restrict__ l, float* __restrict__ red, int E) {
  __shared__ float sh[256];
  int i = blockIdx.x * 256 + threadIdx.x;
  sh[threadIdx.x] = (i < E) ? expf(l[i] - red[0]) : 0.f;
  __syncthreads();
  for (int st = 128; st > 0; st >>= 1) {
    if (threadIdx.x < st) sh[threadIdx.x] += sh[threadIdx.x + st];
    __syncthreads();
  }
  if (threadIdx.x == 0) atomicAdd(&red[1], sh[0]);
}

__global__ void __launch_bounds__(256)
k_probs(const float* __restrict__ l, const float* __restrict__ red,
        float* __restrict__ probs, float* __restrict__ ent, int E) {
  __shared__ float sh[256];
  int i = blockIdx.x * 256 + threadIdx.x;
  float c = 0.f;
  if (i < E) {
    float p = expf(l[i] - red[0]) / red[1];
    probs[i] = p;
    c = -p * logf(p + 1e-10f);
  }
  sh[threadIdx.x] = c; __syncthreads();
  for (int st = 128; st > 0; st >>= 1) {
    if (threadIdx.x < st) sh[threadIdx.x] += sh[threadIdx.x + st];
    __syncthreads();
  }
  if (threadIdx.x == 0) atomicAdd(ent, sh[0]);
}

__global__ void __launch_bounds__(256)
k_sv(const float* __restrict__ av, const float* __restrict__ p,
     const unsigned char* __restrict__ am, float* __restrict__ sv, int E) {
  __shared__ float sh[256];
  int i = blockIdx.x * 256 + threadIdx.x;
  float c = 0.f;
  if (i < E && am[i]) c = av[i] * p[i];
  sh[threadIdx.x] = c; __syncthreads();
  for (int st = 128; st > 0; st >>= 1) {
    if (threadIdx.x < st) sh[threadIdx.x] += sh[threadIdx.x + st];
    __syncthreads();
  }
  if (threadIdx.x == 0) atomicAdd(sv, sh[0]);
}

// ---------------------------------------------------------------------------
// Host launcher
// ---------------------------------------------------------------------------
struct LinP { const float* w; const float* b; };
struct GnP  { const float* w; const float* b; const float* ms; };
struct GatP { const float* as; const float* ad; const float* bias; const float* w; };
struct ConvP { LinP comb0, comb1; GatP gat; GnP gn; LinP trans0, trans1; };

extern "C" void kernel_launch(void* const* d_in, const int* in_sizes, int n_in,
                              void* d_out, int out_size, void* d_ws, size_t ws_size,
                              hipStream_t stream) {
  (void)in_sizes; (void)n_in; (void)out_size; (void)ws_size;
  // ---- unpack params in JAX pytree (alphabetical) leaf order ----
  const float* P = (const float*)d_in[0];
  size_t po = 0;
  auto take = [&](size_t n) { const float* p = P + po; po += n; return p; };
  ConvP conv[2];
  for (int i = 0; i < 2; ++i) {
    conv[i].comb0.b = take(HH); conv[i].comb0.w = take(2 * HH * HH);
    conv[i].comb1.b = take(HH); conv[i].comb1.w = take(2 * HH * HH);
    conv[i].gat.ad  = take(HH); conv[i].gat.as  = take(HH);
    conv[i].gat.bias = take(HH); conv[i].gat.w  = take(HH * HH);
    conv[i].gn.b = take(HH); conv[i].gn.ms = take(HH); conv[i].gn.w = take(HH);
    conv[i].trans0.b = take(HH); conv[i].trans0.w = take(HH * HH);
    conv[i].trans1.b = take(HH); conv[i].trans1.w = take(HH * HH);
  }
  LinP edge_in; edge_in.b = take(HH); edge_in.w = take(DD * HH);
  GnP gns[2];
  for (int i = 0; i < 2; ++i) { gns[i].b = take(HH); gns[i].ms = take(HH); gns[i].w = take(HH); }
  LinP node_in; node_in.b = take(HH); node_in.w = take(DD * HH);
  LinP nqm;  nqm.b  = take(HH); nqm.w  = take(2 * HH * HH);
  LinP pol0; pol0.b = take(HH); pol0.w = take(HH * HH);
  LinP pol1; pol1.b = take(1);  pol1.w = take(HH);
  LinP q_in; q_in.b = take(HH); q_in.w = take(DD * HH);
  LinP tm0;  tm0.b  = take(HH); tm0.w  = take(3 * HH * HH);
  LinP tm1;  tm1.b  = take(HH); tm1.w  = take(HH * HH);
  LinP val0; val0.b = take(HH); val0.w = take(HH * HH);
  LinP val1; val1.b = take(1);  val1.w = take(HH);

  const float* x_    = (const float*)d_in[1];
  const int*   ei    = (const int*)d_in[2];
  const int*   src   = ei;
  const int*   dst   = ei + EE;
  const float* eattr = (const float*)d_in[3];
  const float* qemb  = (const float*)d_in[4];
  const unsigned char* smask = (const unsigned char*)d_in[5];
  const unsigned char* amask = (const unsigned char*)d_in[6];
  const float* abias = (const float*)d_in[7];

  // ---- workspace layout ----
  char* w8 = (char*)d_ws;
  size_t wo = 0;
  auto wal = [&](size_t nfl) { float* p = (float*)(w8 + wo); wo += nfl * sizeof(float); return p; };
  float* nA = wal((size_t)NN * HH);
  float* nB = wal((size_t)NN * HH);
  float* nC = wal((size_t)NN * HH);
  float* nD = wal((size_t)NN * HH);
  float* eEb = wal((size_t)EE * HH);   // e, later reused
  float* eFb = wal((size_t)EE * HH);   // triple hidden / head hidden
  float* qv = wal(HH);
  float* colsum = wal(HH); float* meanb = wal(HH); float* rstdb = wal(HH);
  float* ssum = wal(NN); float* dsum = wal(NN);
  float* mmax = wal(NN); float* denom = wal(NN);
  float* aE = wal(EE); float* eaE = wal(EE);
  float* logit = wal(EE); float* av = wal(EE);
  float* red = wal(2);

  float* o    = (float*)d_out;
  float* oP   = o;
  float* oSV  = o + EE;
  float* oT   = o + EE + 1;
  float* oEnt = o + EE + 1 + (size_t)EE * HH;

  auto gemm = [&](int mode, int M, int K, int lda, const float* A0, const float* A1,
                  const int* s, const int* d, const float* W, const float* B,
                  int relu, float* out) {
    dim3 grid((M + 127) / 128, HH / 64);
    switch (mode) {
      case 0: k_gemm<0><<<grid, 256, 0, stream>>>(M, K, lda, A0, A1, s, d, W, B, relu, out); break;
      case 1: k_gemm<1><<<grid, 256, 0, stream>>>(M, K, lda, A0, A1, s, d, W, B, relu, out); break;
      case 2: k_gemm<2><<<grid, 256, 0, stream>>>(M, K, lda, A0, A1, s, d, W, B, relu, out); break;
      default: k_gemm<3><<<grid, 256, 0, stream>>>(M, K, lda, A0, A1, s, d, W, B, relu, out); break;
    }
  };
  auto fill = [&](float* p, float v, size_t n) {
    k_fill<<<(unsigned)((n + 255) / 256), 256, 0, stream>>>(p, v, n);
  };
  auto gnrun = [&](const float* in, float* out, const GnP& g, int M) {
    fill(colsum, 0.f, HH);
    k_colsum<<<(M + 63) / 64, 256, 0, stream>>>(in, colsum, M);
    k_finalize<<<1, 256, 0, stream>>>(colsum, meanb, (float)M, 0.f, 0);
    fill(colsum, 0.f, HH);
    k_colvar<<<(M + 63) / 64, 256, 0, stream>>>(in, meanb, g.ms, colsum, M);
    k_finalize<<<1, 256, 0, stream>>>(colsum, rstdb, (float)M, 1e-5f, 1);
    int tot = M * HH;
    k_gn_apply<<<(tot + 255) / 256, 256, 0, stream>>>(in, meanb, rstdb, g.ms, g.w, g.b, out, M);
  };

  // ---- input projections ----
  gemm(0, 1,  DD, DD, qemb,  nullptr, nullptr, nullptr, q_in.w,    q_in.b,    1, qv);
  gemm(0, NN, DD, DD, x_,    nullptr, nullptr, nullptr, node_in.w, node_in.b, 1, nA);
  gemm(0, EE, DD, DD, eattr, nullptr, nullptr, nullptr, edge_in.w, edge_in.b, 1, eEb);
  gemm(1, NN, 2 * HH, 0, nA, qv, nullptr, nullptr, nqm.w, nqm.b, 1, nB);  // x in nB

  // ---- two GLASS conv layers ----
  for (int L = 0; L < 2; ++L) {
    const ConvP& c = conv[L];
    gemm(0, NN, HH, HH, nB, nullptr, nullptr, nullptr, c.trans1.w, c.trans1.b, 1, nA);
    gemm(0, NN, HH, HH, nB, nullptr, nullptr, nullptr, c.trans0.w, c.trans0.b, 1, nC);
    { int tot = NN * HH; k_mix<<<(tot + 255) / 256, 256, 0, stream>>>(smask, nA, nC, nA, NN); }
    gemm(0, NN, HH, HH, nA, nullptr, nullptr, nullptr, c.gat.w, nullptr, 0, nC);  // h
    k_rowatt<<<(NN + 7) / 8, 256, 0, stream>>>(nC, c.gat.as, c.gat.ad, ssum, dsum, NN);
    fill(mmax, -INFINITY, NN); fill(denom, 0.f, NN); fill(nD, 0.f, (size_t)NN * HH);
    k_edge_a_max<<<(EE + 255) / 256, 256, 0, stream>>>(ssum, dsum, src, dst, aE, mmax, EE);
    k_node_self_max<<<(NN + 255) / 256, 256, 0, stream>>>(ssum, dsum, mmax, NN);
    k_edge_exp<<<(EE + 255) / 256, 256, 0, stream>>>(aE, mmax, dst, eaE, denom, EE);
    k_node_self_exp<<<(NN + 255) / 256, 256, 0, stream>>>(ssum, dsum, mmax, denom, NN);
    k_edge_scatter<<<EE, 256, 0, stream>>>(eaE, denom, src, dst, nC, nD, EE);
    k_node_self_scatter<<<NN, 256, 0, stream>>>(ssum, dsum, mmax, denom, nC, nD);
    { int tot = NN * HH; k_add_bias<<<(tot + 255) / 256, 256, 0, stream>>>(nD, c.gat.bias, tot); }
    gnrun(nD, nA, c.gn, NN);
    gemm(2, NN, 2 * HH, 0, nA, nB, nullptr, nullptr, c.comb1.w, c.comb1.b, 0, nC);
    gemm(2, NN, 2 * HH, 0, nA, nB, nullptr, nullptr, c.comb0.w, c.comb0.b, 0, nD);
    { int tot = NN * HH; k_mix<<<(tot + 255) / 256, 256, 0, stream>>>(smask, nC, nD, nC, NN); }
    gnrun(nC, nB, gns[L], NN);  // x back in nB
  }

  // ---- triple mix -> t ----
  gemm(3, EE, 3 * HH, 0, nB, eEb, src, dst, tm0.w, tm0.b, 1, eFb);
  gemm(0, EE, HH, HH, eFb, nullptr, nullptr, nullptr, tm1.w, tm1.b, 0, oT);

  // ---- policy head ----
  gemm(0, EE, HH, HH, oT, nullptr, nullptr, nullptr, pol0.w, pol0.b, 1, eFb);
  k_dot_head<<<(EE + 7) / 8, 256, 0, stream>>>(eFb, pol1.w, pol1.b, logit, EE);
  k_logit_post<<<(EE + 255) / 256, 256, 0, stream>>>(logit, amask, abias, EE);
  fill(red, -INFINITY, 1); fill(red + 1, 0.f, 1);
  k_redmax<<<(EE + 255) / 256, 256, 0, stream>>>(logit, red, EE);
  k_redsum<<<(EE + 255) / 256, 256, 0, stream>>>(logit, red, EE);
  fill(oSV, 0.f, 1); fill(oEnt, 0.f, 1);
  k_probs<<<(EE + 255) / 256, 256, 0, stream>>>(logit, red, oP, oEnt, EE);

  // ---- value head ----
  gemm(0, EE, HH, HH, oT, nullptr, nullptr, nullptr, val0.w, val0.b, 1, eFb);
  k_dot_head<<<(EE + 7) / 8, 256, 0, stream>>>(eFb, val1.w, val1.b, av, EE);
  k_sv<<<(EE + 255) / 256, 256, 0, stream>>>(av, oP, amask, oSV, EE);
}
